// DirectEncodingModelTrainable_77077483094889
// MI455X (gfx1250) — matrix operand order repaired and checked by reading, not compiled, not run on hardware
//
#include <hip/hip_runtime.h>
#include <hip/hip_bf16.h>
#include <cstdint>
#include <cstddef>

typedef __attribute__((ext_vector_type(16))) __bf16 v16bf;
typedef __attribute__((ext_vector_type(8)))  __bf16 v8bf;
typedef __attribute__((ext_vector_type(4)))  __bf16 v4bf;
typedef __attribute__((ext_vector_type(8)))  float  v8f;
typedef __attribute__((ext_vector_type(4)))  unsigned int v4u;
typedef __attribute__((ext_vector_type(8)))  int v8i;
typedef __attribute__((ext_vector_type(4)))  int v4i;

#define B_ROWS 8192
#define N_IN   1024
#define WIDTH  1024
#define FAN_IN 1536
#define ACC_W  4096   // max gathered width (idx3 < 4096); bf16 acc = 64 MiB, L2-resident

// ---- feature detection ------------------------------------------------------
#if defined(__has_builtin)
#if __has_builtin(__builtin_amdgcn_tensor_load_to_lds) && __has_builtin(__builtin_amdgcn_s_wait_tensorcnt)
#define USE_TDM 1
#endif
#if __has_builtin(__builtin_amdgcn_tanhf)
#define TANHF(x) __builtin_amdgcn_tanhf(x)
#endif
#endif
#ifndef USE_TDM
#define USE_TDM 0
#endif
#ifndef TANHF
#define TANHF(x) tanhf(x)
#endif

// LDS-only split barrier: avoids draining global LOADcnt at each sync
static __device__ __forceinline__ void block_sync_lds() {
#if __has_builtin(__builtin_amdgcn_s_barrier_signal) && __has_builtin(__builtin_amdgcn_s_barrier_wait)
  __builtin_amdgcn_fence(__ATOMIC_RELEASE, "workgroup", "local");
  __builtin_amdgcn_s_barrier_signal(-1);
  __builtin_amdgcn_s_barrier_wait(-1);
  __builtin_amdgcn_fence(__ATOMIC_ACQUIRE, "workgroup", "local");
#else
  __syncthreads();
#endif
}

// ---------------------------------------------------------------- cvt x -> bf16 acc
__global__ __launch_bounds__(256) void cvt_x_kernel(const float* __restrict__ x,
                                                    __bf16* __restrict__ acc) {
  int t = blockIdx.x * 256 + threadIdx.x;          // over B*N_IN/4
  int m = t >> 8;
  int c = (t & 255) << 2;
  float4 v = reinterpret_cast<const float4*>(x)[t];
  v4bf o = { (__bf16)v.x, (__bf16)v.y, (__bf16)v.z, (__bf16)v.w };
  *reinterpret_cast<v4bf*>(acc + (size_t)m * ACC_W + c) = o;
}

// ---------------------------------------------------------------- K [1536][1024] f32 -> Kp [1024][1536] bf16 (coalesced both sides)
__global__ __launch_bounds__(256) void pack_k_kernel(const float* __restrict__ K,
                                                     __bf16* __restrict__ Kp) {
  __shared__ float tile[32][33];
  int tx = threadIdx.x & 31;
  int ty = threadIdx.x >> 5;                       // 0..7
  int k0 = blockIdx.x * 32;
  int n0 = blockIdx.y * 32;
#pragma unroll
  for (int r = 0; r < 32; r += 8)
    tile[ty + r][tx] = K[(size_t)(k0 + ty + r) * WIDTH + n0 + tx];
  __syncthreads();
#pragma unroll
  for (int r = 0; r < 32; r += 8)
    Kp[(size_t)(n0 + ty + r) * FAN_IN + k0 + tx] = (__bf16)tile[tx][ty + r];
}

// ---------------------------------------------------------------- gather: ag[m][j] = acc[m][idx[j]]  (L2-resident reads)
__global__ __launch_bounds__(256) void gather_kernel(const __bf16* __restrict__ acc,
                                                     const int* __restrict__ idx,
                                                     __bf16* __restrict__ ag) {
  int m = blockIdx.x;
  const __bf16* row = acc + (size_t)m * ACC_W;
  __bf16* orow = ag + (size_t)m * FAN_IN;
  for (int j = threadIdx.x; j < FAN_IN; j += 256)
    orow[j] = row[idx[j]];
}

// ---------------------------------------------------------------- GEMM + bias + tanh
// Block tile 256x128, 8 waves as 4(M) x 2(N); wave tile 64x64:
//   per K-step per wave: 16 ds_load_b128 for 16 WMMAs (ratio 1.0)
#define BM 256
#define BN 128
#define BK 32
#define PADK 40   // 80B row stride: 16B aligned, conflict-free b128 frag loads
#define ABUF (BM * PADK)   // elements per A buffer
#define BBUF (BN * PADK)   // elements per B buffer

union Frag { v16bf v; v8bf h[2]; };

#if USE_TDM
// TDM D#: 2-D tile [tile_rows x 32] bf16 from a row-major tensor (row stride FAN_IN),
// written to LDS with +16B padding per 64B row => PADK=40 layout.
static __device__ __forceinline__ void tdm_load_tile(unsigned lds_addr,
                                                     const __bf16* gptr,
                                                     int tile_rows,
                                                     int tensor_rows) {
  uint64_t ga = (uint64_t)(uintptr_t)gptr;
  v4u g0 = { 1u,                                   // count=1, user descriptor
             lds_addr,                             // LDS byte address
             (unsigned)ga,                         // global_addr[31:0]
             (unsigned)((ga >> 32) & 0x01FFFFFFu) | (2u << 30) };  // addr[56:32] | type=2
  v8i g1 = { (int)((1u << 16)      // data_size = 1 -> 2 bytes
                 | (1u << 20)      // pad_enable
                 | (3u << 22)      // pad_interval = 3 -> every 16 DWORDs (64B)
                 | (3u << 25)),    // pad_amount  = 3 -> 4 DWORDs (16B)
             (int)(((unsigned)FAN_IN & 0xFFFFu) << 16),        // tensor_dim0[15:0]
             (int)(((unsigned)tensor_rows & 0xFFFFu) << 16),   // dim0 hi=0 | tensor_dim1[15:0]
             (int)(32u << 16),                                 // dim1 hi=0 | tile_dim0=32
             (int)tile_rows,                                   // tile_dim1 | tile_dim2=0
             (int)FAN_IN,                                      // tensor_dim0_stride[31:0]
             0, 0 };
  v4i gz = {0, 0, 0, 0};
#if __clang_major__ >= 23
  v8i gz8 = {0, 0, 0, 0, 0, 0, 0, 0};
  __builtin_amdgcn_tensor_load_to_lds(g0, g1, gz, gz, gz8, 0);
#else
  __builtin_amdgcn_tensor_load_to_lds(g0, g1, gz, gz, 0);
#endif
}
#endif

template <bool LAST>
__global__ __launch_bounds__(256)
__attribute__((amdgpu_num_vgpr(256)))   // guardrail: stay in the directly-addressable window
void gemm_tanh_kernel(
    const __bf16* __restrict__ A,    // [8192][1536] gathered activations
    const __bf16* __restrict__ Bp,   // [1024][1536] n-major weights
    const float*  __restrict__ bias, // [1024]
    __bf16* __restrict__ out_bf,     // acc + col offset (stride ACC_W), LAST==false
    float*  __restrict__ out_f)      // [8192][1024], LAST==true
{
  __shared__ __bf16 sa[2 * ABUF];   // 40 KB
  __shared__ __bf16 sb[2 * BBUF];   // 20 KB

  const int tid  = threadIdx.x;
  const int lane = tid & 31;
  const int wave = tid >> 5;
  const int wm = (wave >> 1) * 64;  // wave M offset in block tile (0,64,128,192)
  const int wn = (wave & 1) * 64;   // wave N offset in block tile (0,64)

  const int mBase = blockIdx.x * BM;
  const int nBase = blockIdx.y * BN;

  v8f accf[4][4];
  const v8f vzero = {0.f, 0.f, 0.f, 0.f, 0.f, 0.f, 0.f, 0.f};
#pragma unroll
  for (int i = 0; i < 4; ++i)
#pragma unroll
    for (int j = 0; j < 4; ++j) accf[i][j] = vzero;

  const int nK = FAN_IN / BK;  // 48

#if USE_TDM
  const __bf16* gAt = A  + (size_t)mBase * FAN_IN;  // tile base (add kt per stage)
  const __bf16* gBt = Bp + (size_t)nBase * FAN_IN;
  const unsigned lds_a0 = (unsigned)(uintptr_t)&sa[0];
  const unsigned lds_b0 = (unsigned)(uintptr_t)&sb[0];

  if (wave == 0) {
    tdm_load_tile(lds_a0, gAt, BM, B_ROWS);
    tdm_load_tile(lds_b0, gBt, BN, WIDTH);
    __builtin_amdgcn_s_wait_tensorcnt(0);
  }
  block_sync_lds();
#else
  // ---- register staging fallback
  const int lcol = (tid & 3) * 8;   // 0,8,16,24
  const int lrow = tid >> 2;        // 0..63
  const __bf16* gA0 = A  + (size_t)(mBase + lrow)       * FAN_IN + lcol;
  const __bf16* gA1 = A  + (size_t)(mBase + lrow + 64)  * FAN_IN + lcol;
  const __bf16* gA2 = A  + (size_t)(mBase + lrow + 128) * FAN_IN + lcol;
  const __bf16* gA3 = A  + (size_t)(mBase + lrow + 192) * FAN_IN + lcol;
  const __bf16* gB0 = Bp + (size_t)(nBase + lrow)       * FAN_IN + lcol;
  const __bf16* gB1 = Bp + (size_t)(nBase + lrow + 64)  * FAN_IN + lcol;
  const int sOff0 = lrow * PADK + lcol;
  const int sOff1 = (lrow + 64)  * PADK + lcol;
  const int sOff2 = (lrow + 128) * PADK + lcol;
  const int sOff3 = (lrow + 192) * PADK + lcol;

  uint4 ra0 = *reinterpret_cast<const uint4*>(gA0);
  uint4 ra1 = *reinterpret_cast<const uint4*>(gA1);
  uint4 ra2 = *reinterpret_cast<const uint4*>(gA2);
  uint4 ra3 = *reinterpret_cast<const uint4*>(gA3);
  uint4 rb0 = *reinterpret_cast<const uint4*>(gB0);
  uint4 rb1 = *reinterpret_cast<const uint4*>(gB1);
  *reinterpret_cast<uint4*>(&sa[sOff0]) = ra0;
  *reinterpret_cast<uint4*>(&sa[sOff1]) = ra1;
  *reinterpret_cast<uint4*>(&sa[sOff2]) = ra2;
  *reinterpret_cast<uint4*>(&sa[sOff3]) = ra3;
  *reinterpret_cast<uint4*>(&sb[sOff0]) = rb0;
  *reinterpret_cast<uint4*>(&sb[sOff1]) = rb1;
  block_sync_lds();
  ra0 = *reinterpret_cast<const uint4*>(gA0 + BK);
  ra1 = *reinterpret_cast<const uint4*>(gA1 + BK);
  ra2 = *reinterpret_cast<const uint4*>(gA2 + BK);
  ra3 = *reinterpret_cast<const uint4*>(gA3 + BK);
  rb0 = *reinterpret_cast<const uint4*>(gB0 + BK);
  rb1 = *reinterpret_cast<const uint4*>(gB1 + BK);
#endif

  const int abase = (wm + (lane & 15)) * PADK + (lane >> 4) * 16;
  const int bbase = (wn + (lane & 15)) * PADK + (lane >> 4) * 16;

  // branchless double-buffer toggles: keeps the loop single-copy (no x2 unroll),
  // so fragment lifetimes stay within one iteration -> no spills / copy chains
  int acur = 0;
  int bcur = 0;

#pragma unroll 1
  for (int ks = 0; ks < nK; ++ks) {
    const __bf16* sac = &sa[0] + (acur + abase);
    const __bf16* sbc = &sb[0] + (bcur + bbase);

#if USE_TDM
    // wave0 kicks off the DMA for the next stage; it lands in the other buffer,
    // which no wave has touched since the barrier at the end of iter ks-1.
    if (wave == 0 && ks + 1 < nK) {
      const int kt = (ks + 1) * BK;
      tdm_load_tile(lds_a0 + (unsigned)(acur ^ ABUF) * 2u, gAt + kt, BM, B_ROWS);
      tdm_load_tile(lds_b0 + (unsigned)(bcur ^ BBUF) * 2u, gBt + kt, BN, WIDTH);
    }
#endif

    // B fragments first (kept live), A fragments streamed
    Frag bfr0, bfr1, bfr2, bfr3;
    bfr0.h[0] = *reinterpret_cast<const v8bf*>(&sbc[0]);
    bfr0.h[1] = *reinterpret_cast<const v8bf*>(&sbc[8]);
    bfr1.h[0] = *reinterpret_cast<const v8bf*>(&sbc[16 * PADK]);
    bfr1.h[1] = *reinterpret_cast<const v8bf*>(&sbc[16 * PADK + 8]);
    bfr2.h[0] = *reinterpret_cast<const v8bf*>(&sbc[32 * PADK]);
    bfr2.h[1] = *reinterpret_cast<const v8bf*>(&sbc[32 * PADK + 8]);
    bfr3.h[0] = *reinterpret_cast<const v8bf*>(&sbc[48 * PADK]);
    bfr3.h[1] = *reinterpret_cast<const v8bf*>(&sbc[48 * PADK + 8]);

#pragma unroll
    for (int mf = 0; mf < 4; ++mf) {
      Frag af;
      af.h[0] = *reinterpret_cast<const v8bf*>(&sac[mf * 16 * PADK]);
      af.h[1] = *reinterpret_cast<const v8bf*>(&sac[mf * 16 * PADK + 8]);
      accf[mf][0] = __builtin_amdgcn_wmma_f32_16x16x32_bf16(false, af.v, false, bfr0.v, (short)0, accf[mf][0], false, false);
      accf[mf][1] = __builtin_amdgcn_wmma_f32_16x16x32_bf16(false, af.v, false, bfr1.v, (short)0, accf[mf][1], false, false);
      accf[mf][2] = __builtin_amdgcn_wmma_f32_16x16x32_bf16(false, af.v, false, bfr2.v, (short)0, accf[mf][2], false, false);
      accf[mf][3] = __builtin_amdgcn_wmma_f32_16x16x32_bf16(false, af.v, false, bfr3.v, (short)0, accf[mf][3], false, false);
    }

#if USE_TDM
    if (wave == 0 && ks + 1 < nK) __builtin_amdgcn_s_wait_tensorcnt(0);
#else
    if (ks + 1 < nK) {
      __bf16* san = &sa[0] + (acur ^ ABUF);
      __bf16* sbn = &sb[0] + (bcur ^ BBUF);
      *reinterpret_cast<uint4*>(&san[sOff0]) = ra0;
      *reinterpret_cast<uint4*>(&san[sOff1]) = ra1;
      *reinterpret_cast<uint4*>(&san[sOff2]) = ra2;
      *reinterpret_cast<uint4*>(&san[sOff3]) = ra3;
      *reinterpret_cast<uint4*>(&sbn[sOff0]) = rb0;
      *reinterpret_cast<uint4*>(&sbn[sOff1]) = rb1;
      if (ks + 2 < nK) {
        const int off = (ks + 2) * BK;
        ra0 = *reinterpret_cast<const uint4*>(gA0 + off);
        ra1 = *reinterpret_cast<const uint4*>(gA1 + off);
        ra2 = *reinterpret_cast<const uint4*>(gA2 + off);
        ra3 = *reinterpret_cast<const uint4*>(gA3 + off);
        rb0 = *reinterpret_cast<const uint4*>(gB0 + off);
        rb1 = *reinterpret_cast<const uint4*>(gB1 + off);
      }
    }
#endif
    acur ^= ABUF;
    bcur ^= BBUF;
    block_sync_lds();
  }

  // epilogue: C frag layout — lane%16 = N col, VGPR r -> M row r + 8*(lane>>4)
#pragma unroll
  for (int nf = 0; nf < 4; ++nf) {
    const int n  = nBase + wn + nf * 16 + (lane & 15);
    const float bv = bias[n];
#pragma unroll
    for (int mf = 0; mf < 4; ++mf) {
      const int m0 = mBase + wm + mf * 16 + (lane >> 4) * 8;
#pragma unroll
      for (int r = 0; r < 8; ++r) {
        float v = TANHF(accf[mf][nf][r] + bv);
        if (LAST) out_f [(size_t)(m0 + r) * WIDTH + n] = v;
        else      out_bf[(size_t)(m0 + r) * ACC_W + n] = (__bf16)v;
      }
    }
  }
}

// ---------------------------------------------------------------- launch
extern "C" void kernel_launch(void* const* d_in, const int* in_sizes, int n_in,
                              void* d_out, int out_size, void* d_ws, size_t ws_size,
                              hipStream_t stream) {
  (void)in_sizes; (void)n_in; (void)out_size; (void)ws_size;
  const float* x = (const float*)d_in[0];

  char* ws = (char*)d_ws;
  const size_t acc_bytes = (size_t)B_ROWS * ACC_W * sizeof(__bf16);    // 64 MiB
  const size_t ag_bytes  = (size_t)B_ROWS * FAN_IN * sizeof(__bf16);   // 24 MiB
  __bf16* acc = (__bf16*)ws;
  __bf16* ag  = (__bf16*)(ws + acc_bytes);
  __bf16* kp  = (__bf16*)(ws + acc_bytes + ag_bytes);                  // 3 MiB

  cvt_x_kernel<<<(B_ROWS * N_IN / 4) / 256, 256, 0, stream>>>(x, acc);

  for (int l = 0; l < 4; ++l) {
    const float* K   = (const float*)d_in[1 + 3 * l];
    const float* b   = (const float*)d_in[2 + 3 * l];
    const int*   idx = (const int*)  d_in[3 + 3 * l];

    pack_k_kernel<<<dim3(FAN_IN / 32, WIDTH / 32), 256, 0, stream>>>(K, kp);
    gather_kernel<<<B_ROWS, 256, 0, stream>>>(acc, idx, ag);

    if (l < 3) {
      __bf16* obf = acc + (size_t)(l + 1) * 1024;
      gemm_tanh_kernel<false><<<dim3(B_ROWS / BM, WIDTH / BN), 256, 0, stream>>>(ag, kp, b, obf, nullptr);
    } else {
      gemm_tanh_kernel<true><<<dim3(B_ROWS / BM, WIDTH / BN), 256, 0, stream>>>(ag, kp, b, nullptr, (float*)d_out);
    }
  }
}
